// TemporalDecoder_38603166056606
// MI455X (gfx1250) — matrix operand order repaired and checked by reading, not compiled, not run on hardware
//
#include <hip/hip_runtime.h>

// Problem dims (fixed by reference)
#define B_  4
#define T_  16
#define N_  128
#define D_  16
#define H_  64
#define MH_ 128

typedef __attribute__((ext_vector_type(16))) _Float16 v16h;
typedef __attribute__((ext_vector_type(8)))  _Float16 v8h;
typedef __attribute__((ext_vector_type(8)))  float    v8f;

// ---------------------------------------------------------------- math helpers
__device__ __forceinline__ float dev_tanhf(float x) {
#if __has_builtin(__builtin_amdgcn_tanhf)
  return __builtin_amdgcn_tanhf(x);          // v_tanh_f32 (TRANS, co-executes)
#elif __has_builtin(__builtin_amdgcn_tanh_f32)
  return __builtin_amdgcn_tanh_f32(x);
#else
  x = fminf(fmaxf(x, -10.f), 10.f);
  float e = __expf(2.f * x);
  return (e - 1.f) / (e + 1.f);
#endif
}
__device__ __forceinline__ _Float16 dev_tanh_h(_Float16 x) {
#if __has_builtin(__builtin_amdgcn_tanh_f16)
  return __builtin_amdgcn_tanh_f16(x);       // v_tanh_f16
#elif __has_builtin(__builtin_amdgcn_tanhh)
  return __builtin_amdgcn_tanhh(x);
#else
  return (_Float16)dev_tanhf((float)x);
#endif
}
__device__ __forceinline__ float fast_sigmoid(float x) {
  return 1.f / (1.f + __expf(-x));
}

// ---------------------------------------------------------------- WMMA helpers
// A operand (f16, 16x32, row major src [16][ld], ld multiple of 8):
//   lane<16: row=lane,   K = k0+{0..7} then k0+{16..23}
//   lane>=16: row=lane-16, K = k0+{8..15} then k0+{24..31}
__device__ __forceinline__ v16h load_a16(const _Float16* src, int ld, int k0) {
  int lane = threadIdx.x & 31;
  int row  = lane & 15;
  int kb   = k0 + ((lane & 16) ? 8 : 0);
  v8h lo = *(const v8h*)&src[row * ld + kb];        // 16B vector load
  v8h hi = *(const v8h*)&src[row * ld + kb + 16];   // 16B vector load
  v16h a;
#pragma unroll
  for (int e = 0; e < 8; ++e) { a[e] = lo[e]; a[8 + e] = hi[e]; }
  return a;
}
// B operand from pre-packed fragment buffer: fragment fid, lane-major layout
//   pack[((fid*32 + lane)*16 + e] == element e of lane's B fragment
__device__ __forceinline__ v16h load_b_packed(const _Float16* pack, int fid) {
  int lane = threadIdx.x & 31;
  const _Float16* p = pack + ((size_t)fid * 32 + lane) * 16;
  v8h lo = *(const v8h*)p;          // global_load_b128, coalesced
  v8h hi = *(const v8h*)(p + 8);    // global_load_b128, coalesced
  v16h b;
#pragma unroll
  for (int e = 0; e < 8; ++e) { b[e] = lo[e]; b[8 + e] = hi[e]; }
  return b;
}
// C/D f32 16x16: VGPR v -> row v + 8*(lane>=16), col = lane&15 ; store as f16
__device__ __forceinline__ void store_c_f16(_Float16* dst, int ld, v8f c) {
  int lane = threadIdx.x & 31;
  int col  = lane & 15;
  int rb   = (lane & 16) ? 8 : 0;
#pragma unroll
  for (int v = 0; v < 8; ++v) dst[(rb + v) * ld + col] = (_Float16)c[v];
}

// ---------------------------------------------------------------- prep kernel
// Packs W_msg1 / W_msg2 into WMMA B-fragment order (f16), symmetrizes A,
// zeroes hidden state and the loss accumulator.
__global__ __launch_bounds__(256) void prep_kernel(
    const float* __restrict__ A, const float* __restrict__ W_msg1,
    const float* __restrict__ W_msg2,
    float* __restrict__ Asym, _Float16* __restrict__ W1p, _Float16* __restrict__ W2p,
    float* __restrict__ hidden0, _Float16* __restrict__ hiddenh0,
    float* __restrict__ loss) {
  int idx = blockIdx.x * blockDim.x + threadIdx.x;
  if (idx < B_ * N_ * N_) {
    int b = idx >> 14, i = (idx >> 7) & 127, j = idx & 127;
    float v = 0.5f * (A[(b * N_ + i) * N_ + j] + A[(b * N_ + j) * N_ + i]);
    if (i == j) v = 0.f;
    Asym[idx] = fminf(fmaxf(v, 0.f), 1.f);
  }
  if (idx < 32 * 32 * 16) {   // W1 fragments: fid = z*16 + kt*8 + nt (32 frags)
    int e = idx & 15, l = (idx >> 4) & 31, fid = idx >> 9;
    int z = fid >> 4, kt = (fid >> 3) & 1, nt = fid & 7;
    int col = (l & 15) + nt * 16;
    int kk  = kt * 32 + ((l & 16) ? 8 : 0) + (e < 8 ? e : e + 8);      // K: table order (8h+e, then 16+8h+e-8)
    W1p[idx] = (_Float16)W_msg1[(z * H_ + kk) * MH_ + col];
  }
  if (idx < 16 * 32 * 16) {   // W2 fragments: fid = kt*4 + nt (16 frags)
    int e = idx & 15, l = (idx >> 4) & 31, fid = idx >> 9;
    int kt = fid >> 2, nt = fid & 3;
    int col = (l & 15) + nt * 16;
    int kk  = kt * 32 + ((l & 16) ? 8 : 0) + (e < 8 ? e : e + 8);
    W2p[idx] = (_Float16)W_msg2[kk * H_ + col];
  }
  if (idx < B_ * N_ * H_) { hidden0[idx] = 0.f; hiddenh0[idx] = (_Float16)0.f; }
  if (idx == 0) loss[0] = 0.f;
}

// ------------------------------------------------- per-step projection (WMMA)
// hs = hidden @ W1s ; hr = hidden @ W1r + b_msg1 (f16 outputs); zero agg (z==0)
__global__ __launch_bounds__(32) void proj_kernel(
    const _Float16* __restrict__ hiddenh,  // (B*N=512, H=64) f16
    const _Float16* __restrict__ W1p,      // packed fragments
    const float* __restrict__ b_msg1,      // (128)
    _Float16* __restrict__ hs, _Float16* __restrict__ hr,  // (512, 128) f16
    float* __restrict__ agg)               // (B*N, H) f32, zeroed here
{
  int rt = blockIdx.x;   // 0..31 row tile (16 rows of B*N)
  int nt = blockIdx.y;   // 0..7  col tile (16 cols of MH)
  int z  = blockIdx.z;   // 0: hs, 1: hr
  if (z == 0) {          // 256 z==0 blocks x 128 floats = 32768 = B*N*H
    int base = (rt * 8 + nt) * 128;
    for (int e = threadIdx.x; e < 128; e += 32) agg[base + e] = 0.f;
  }
  const _Float16* arow = hiddenh + rt * 16 * H_;
  v16h a0 = load_a16(arow, H_, 0);
  v16h a1 = load_a16(arow, H_, 32);
  v16h b0 = load_b_packed(W1p, z * 16 + nt);        // kt=0
  v16h b1 = load_b_packed(W1p, z * 16 + 8 + nt);    // kt=1
  v8f c = {};
  if (z == 1) {
    float bb = b_msg1[nt * 16 + (threadIdx.x & 15)];
#pragma unroll
    for (int v = 0; v < 8; ++v) c[v] = bb;
  }
  c = __builtin_amdgcn_wmma_f32_16x16x32_f16(false, a0, false, b0, (short)0, c, false, false);
  c = __builtin_amdgcn_wmma_f32_16x16x32_f16(false, a1, false, b1, (short)0, c, false, false);
  _Float16* dst = (z ? hr : hs) + rt * 16 * MH_ + nt * 16;
  store_c_f16(dst, MH_, c);
}

// --------------------------------------- fused message GEMM + aggregate (WMMA)
// For block (b, j-tile, i-tile):
//   m1[i,j,:] = tanh(hs[i,:] + hr[j,:])   (b128 LDS reads + v_pk_add_f16 + v_tanh_f16)
//   msg       = tanh(m1 @ W_msg2 + b_msg2)            (WMMA f16 -> f32)
//   agg[b,j,:] += sum_i Asym[b,i,j] * msg[i,:]         (f32 atomics)
__global__ __launch_bounds__(128) void msg_agg_kernel(
    const _Float16* __restrict__ hs, const _Float16* __restrict__ hr,
    const float* __restrict__ Asym, const _Float16* __restrict__ W2p,
    const float* __restrict__ b_msg2, float* __restrict__ agg)
{
  __shared__ __align__(16) _Float16 hs_s[16][MH_];
  __shared__ __align__(16) _Float16 hr_s[16][MH_];
  __shared__ float a_s[16][16];
  int it = blockIdx.x, jt = blockIdx.y, b = blockIdx.z;
  int i0 = it * 16, j0 = jt * 16;
  // Stage f16 tiles with 16B vector copies: 4096B per tile = 256 uint4
  const uint4* hsrc  = (const uint4*)(hs + (b * N_ + i0) * MH_);
  const uint4* hrsrc = (const uint4*)(hr + (b * N_ + j0) * MH_);
  uint4* hss = (uint4*)&hs_s[0][0];
  uint4* hrs = (uint4*)&hr_s[0][0];
  for (int idx = threadIdx.x; idx < 16 * MH_ / 8; idx += 128) {
    hss[idx] = hsrc[idx];
    hrs[idx] = hrsrc[idx];
  }
  for (int idx = threadIdx.x; idx < 256; idx += 128)
    a_s[idx >> 4][idx & 15] = Asym[(b * N_ + i0 + (idx >> 4)) * N_ + j0 + (idx & 15)];
  __syncthreads();

  int wave = threadIdx.x >> 5;
  int lane = threadIdx.x & 31;
  int col  = lane & 15;
  int rb   = (lane & 16) ? 8 : 0;
  int row  = lane & 15;

  // W_msg2 fragments: K=128 -> 4 k-tiles, N=64 -> 4 n-tiles; kept in VGPRs.
  v16h bf[4][4];
#pragma unroll
  for (int kt = 0; kt < 4; ++kt)
#pragma unroll
    for (int nt = 0; nt < 4; ++nt)
      bf[kt][nt] = load_b_packed(W2p, kt * 4 + nt);

  float bias[4];
#pragma unroll
  for (int nt = 0; nt < 4; ++nt) bias[nt] = b_msg2[nt * 16 + col];

  for (int jj = wave; jj < 16; jj += 4) {
    const _Float16* hrj = hr_s[jj];
    v8f c[4];
#pragma unroll
    for (int nt = 0; nt < 4; ++nt)
#pragma unroll
      for (int v = 0; v < 8; ++v) c[nt][v] = bias[nt];

#pragma unroll
    for (int kt = 0; kt < 4; ++kt) {
      // Build A fragment: two 16B LDS reads per operand half,
      // packed f16 add + native f16 tanh
      int kb = kt * 32 + ((lane & 16) ? 8 : 0);
      v8h s0 = *(const v8h*)&hs_s[row][kb];
      v8h s1 = *(const v8h*)&hs_s[row][kb + 16];
      v8h r0 = *(const v8h*)&hrj[kb];
      v8h r1 = *(const v8h*)&hrj[kb + 16];
      v8h m0 = s0 + r0;   // 4x v_pk_add_f16
      v8h m1 = s1 + r1;
      v16h a;
#pragma unroll
      for (int e = 0; e < 8; ++e) {
        a[e]     = dev_tanh_h(m0[e]);
        a[8 + e] = dev_tanh_h(m1[e]);
      }
#pragma unroll
      for (int nt = 0; nt < 4; ++nt)
        c[nt] = __builtin_amdgcn_wmma_f32_16x16x32_f16(false, a, false, bf[kt][nt],
                                                       (short)0, c[nt], false, false);
    }
    // msg = tanh(c); weight rows by Asym[b, i, j]; reduce over the 16 i-rows
    float wgt[8];
#pragma unroll
    for (int v = 0; v < 8; ++v) wgt[v] = a_s[rb + v][jj];
#pragma unroll
    for (int nt = 0; nt < 4; ++nt) {
      float s = 0.f;
#pragma unroll
      for (int v = 0; v < 8; ++v)
        s += wgt[v] * dev_tanhf(c[nt][v]);
      s += __shfl_xor(s, 16, 32);   // combine row halves (lanes L and L+16)
      if (lane < 16)
        atomicAdd(&agg[(b * N_ + j0 + jj) * H_ + nt * 16 + col], s);
    }
  }
}

// ----------------------------------------- GRU gate + output MLP + loss kernel
__global__ __launch_bounds__(64) void gru_out_kernel(
    const float* __restrict__ X, const float* __restrict__ agg,
    const float* __restrict__ h_in, float* __restrict__ h_out,
    _Float16* __restrict__ h_out_h,
    const float* __restrict__ W_ir, const float* __restrict__ b_ir,
    const float* __restrict__ W_ii, const float* __restrict__ b_ii,
    const float* __restrict__ W_in, const float* __restrict__ b_in,
    const float* __restrict__ W_hr, const float* __restrict__ W_hi,
    const float* __restrict__ W_hh,
    const float* __restrict__ W_o1, const float* __restrict__ b_o1,
    const float* __restrict__ W_o2, const float* __restrict__ b_o2,
    const float* __restrict__ W_o3, const float* __restrict__ b_o3,
    const float* __restrict__ log_sigma,
    float* __restrict__ pred_out, float* __restrict__ loss, int t)
{
  int bn = blockIdx.x;          // 0..511 = (b,n)
  int b = bn >> 7, n = bn & 127;
  int h = threadIdx.x;          // 0..63
  __shared__ float x_s[D_], agg_s[H_], hn_s[H_], p1_s[H_], p2_s[H_], red[64];

  const float* xrow = X + (((size_t)b * T_ + t) * N_ + n) * D_;
  if (h < D_) x_s[h] = xrow[h];
  agg_s[h] = agg[bn * H_ + h];
  __syncthreads();

  float accr = b_ir[h], acci = b_ii[h], accn = b_in[h];
#pragma unroll
  for (int k = 0; k < D_; ++k) {
    float xv = x_s[k];
    accr += xv * W_ir[k * H_ + h];
    acci += xv * W_ii[k * H_ + h];
    accn += xv * W_in[k * H_ + h];
  }
  float ar = 0.f, ai = 0.f, ah = 0.f;
  for (int k = 0; k < H_; ++k) {
    float av = agg_s[k];
    ar += av * W_hr[k * H_ + h];
    ai += av * W_hi[k * H_ + h];
    ah += av * W_hh[k * H_ + h];
  }
  float r  = fast_sigmoid(accr + ar);
  float ig = fast_sigmoid(acci + ai);
  float nn = dev_tanhf(accn + r * ah);
  float hnew = (1.f - ig) * nn + ig * h_in[bn * H_ + h];
  h_out[bn * H_ + h]   = hnew;
  h_out_h[bn * H_ + h] = (_Float16)hnew;
  hn_s[h] = hnew;
  __syncthreads();

  float p1 = b_o1[h];
  for (int k = 0; k < H_; ++k) p1 += hn_s[k] * W_o1[k * H_ + h];
  p1_s[h] = fmaxf(p1, 0.f);
  __syncthreads();
  float p2 = b_o2[h];
  for (int k = 0; k < H_; ++k) p2 += p1_s[k] * W_o2[k * H_ + h];
  p2_s[h] = fmaxf(p2, 0.f);
  __syncthreads();

  float err2 = 0.f;
  if (h < D_) {
    float pd = b_o3[h];
    for (int k = 0; k < H_; ++k) pd += p2_s[k] * W_o3[k * D_ + h];
    pd += x_s[h];
    // X_pred layout (B, T-1, N, D)
    pred_out[(((size_t)b * (T_ - 1) + t) * N_ + n) * D_ + h] = pd;
    float xn = X[(((size_t)b * T_ + (t + 1)) * N_ + n) * D_ + h];
    float diff = xn - pd;
    err2 = diff * diff;
  }
  red[h] = err2;
  __syncthreads();
  if (h == 0) {
    float s = 0.f;
#pragma unroll
    for (int k = 0; k < D_; ++k) s += red[k];
    float sigma = __expf(log_sigma[0]);
    atomicAdd(loss, -s / (2.f * sigma * sigma));
  }
}

__global__ void finalize_kernel(const float* __restrict__ loss,
                                const float* __restrict__ log_sigma,
                                float* __restrict__ out) {
  if (threadIdx.x == 0) {
    float ls = log_sigma[0];
    // const = N*D*B * log(sigma*sqrt(2*pi)); 15 steps
    float cst = (float)(N_ * D_ * B_) * (ls + 0.91893853320467274178f);
    out[0] = loss[0] - 15.f * cst;
  }
}

// ---------------------------------------------------------------------- launch
extern "C" void kernel_launch(void* const* d_in, const int* in_sizes, int n_in,
                              void* d_out, int out_size, void* d_ws, size_t ws_size,
                              hipStream_t stream) {
  const float* A       = (const float*)d_in[0];
  const float* X       = (const float*)d_in[1];
  const float* W_msg1  = (const float*)d_in[2];
  const float* b_msg1  = (const float*)d_in[3];
  const float* W_msg2  = (const float*)d_in[4];
  const float* b_msg2  = (const float*)d_in[5];
  const float* W_ir    = (const float*)d_in[6];
  const float* b_ir    = (const float*)d_in[7];
  const float* W_ii    = (const float*)d_in[8];
  const float* b_ii    = (const float*)d_in[9];
  const float* W_in_   = (const float*)d_in[10];
  const float* b_in_   = (const float*)d_in[11];
  const float* W_hr    = (const float*)d_in[12];
  const float* W_hi    = (const float*)d_in[13];
  const float* W_hh    = (const float*)d_in[14];
  const float* W_o1    = (const float*)d_in[15];
  const float* b_o1    = (const float*)d_in[16];
  const float* W_o2    = (const float*)d_in[17];
  const float* b_o2    = (const float*)d_in[18];
  const float* W_o3    = (const float*)d_in[19];
  const float* b_o3    = (const float*)d_in[20];
  const float* log_sig = (const float*)d_in[21];

  float* pred_out = (float*)d_out;                           // (B,15,N,D) f32
  float* lout     = (float*)d_out + B_ * (T_ - 1) * N_ * D_; // loglik

  // Workspace carve-up (all chunks stay 256B aligned)
  char* w = (char*)d_ws;
  float*     Asym    = (float*)w;      w += (size_t)B_ * N_ * N_ * 4;   // 256 KB
  _Float16*  W1p     = (_Float16*)w;   w += (size_t)32 * 32 * 16 * 2;   //  32 KB
  _Float16*  W2p     = (_Float16*)w;   w += (size_t)16 * 32 * 16 * 2;   //  16 KB
  float*     hidden0 = (float*)w;      w += (size_t)B_ * N_ * H_ * 4;
  float*     hidden1 = (float*)w;      w += (size_t)B_ * N_ * H_ * 4;
  _Float16*  hh0     = (_Float16*)w;   w += (size_t)B_ * N_ * H_ * 2;
  _Float16*  hh1     = (_Float16*)w;   w += (size_t)B_ * N_ * H_ * 2;
  _Float16*  hs      = (_Float16*)w;   w += (size_t)B_ * N_ * MH_ * 2;  // 128 KB
  _Float16*  hr      = (_Float16*)w;   w += (size_t)B_ * N_ * MH_ * 2;  // 128 KB
  float*     agg     = (float*)w;      w += (size_t)B_ * N_ * H_ * 4;   // 128 KB
  float*     loss    = (float*)w;      w += 256;

  prep_kernel<<<(B_ * N_ * N_ + 255) / 256, 256, 0, stream>>>(
      A, W_msg1, W_msg2, Asym, W1p, W2p, hidden0, hh0, loss);

  float*    hcur = hidden0; float*    hnxt = hidden1;
  _Float16* hhc  = hh0;     _Float16* hhn  = hh1;
  for (int t = 0; t < T_ - 1; ++t) {
    proj_kernel<<<dim3(32, 8, 2), 32, 0, stream>>>(hhc, W1p, b_msg1, hs, hr, agg);
    msg_agg_kernel<<<dim3(8, 8, 4), 128, 0, stream>>>(hs, hr, Asym, W2p, b_msg2, agg);
    gru_out_kernel<<<B_ * N_, 64, 0, stream>>>(
        X, agg, hcur, hnxt, hhn,
        W_ir, b_ir, W_ii, b_ii, W_in_, b_in_, W_hr, W_hi, W_hh,
        W_o1, b_o1, W_o2, b_o2, W_o3, b_o3, log_sig, pred_out, loss, t);
    float* tf = hcur; hcur = hnxt; hnxt = tf;
    _Float16* th = hhc; hhc = hhn; hhn = th;
  }
  finalize_kernel<<<1, 32, 0, stream>>>(loss, log_sig, lout);
}
